// ElevationTransformer_NO4_Var_AT_63256278336220
// MI455X (gfx1250) — compile-verified, matrix operand order checked by reading, not run on hardware
//
#include <hip/hip_runtime.h>
#include <hip/hip_bf16.h>

typedef __attribute__((ext_vector_type(16))) _Float16 v16h;
typedef __attribute__((ext_vector_type(4)))  _Float16 v4h;
typedef __attribute__((ext_vector_type(8)))  float    v8f;

#define NB     64
#define BTS    16
#define IMGSZ  224
#define NPX    (IMGSZ*IMGSZ)
#define EMB    192
#define NPATCH 196
#define NTOK   197
#define VDEPTH 12
#define SATD   256

// ---------------- wave32 helpers ----------------
__device__ __forceinline__ float wave_sum(float v) {
#pragma unroll
  for (int off = 16; off; off >>= 1) v += __shfl_xor(v, off, 32);
  return v;
}
__device__ __forceinline__ float wave_max(float v) {
#pragma unroll
  for (int off = 16; off; off >>= 1) v = fmaxf(v, __shfl_xor(v, off, 32));
  return v;
}

// Fragment-order addressing (ISA 7.12.2, 16-bit A 16x32 / B 32x16):
//   lane = (row_or_col & 15) | ((k & 8) << 1)
//   elem = (k & 7) | ((k >> 1) & 8)
// A lane's 16 halves stored contiguously -> v16h operand = 2x b128 load.

// ---------------- weight pre-pack: f32 [KxN] (opt transposed) -> f16 frag order
__global__ __launch_bounds__(256) void k_packB(const float* __restrict__ src,
                                               _Float16* __restrict__ dst,
                                               int K, int N, int trans) {
  int idx = blockIdx.x * 256 + threadIdx.x;
  if (idx >= K * N) return;
  int k = idx / N, n = idx % N;
  float v = trans ? src[(long long)n * K + k] : src[(long long)k * N + n];
  int kk = k & 31, kstep = k >> 5, nt = n >> 4;
  int lane = (n & 15) | ((kk & 8) << 1);
  int e = (kk & 7) | ((kk >> 1) & 8);
  dst[(((long long)kstep * (N >> 4) + nt) * 32 + lane) * 16 + e] = (_Float16)v;
}

// ---------------- generic batched WMMA GEMM (specialized epilogues) ----------
// C[bz](MxN) = A[bz](MxK f32, ping-pong LDS-staged f16) @ Bp (packed f16 frags)
//              + bias*biasScale  [GELU]  [+= C]
template <int ACC, int GELU>
__global__ __launch_bounds__(256) void k_gemm(
    const float* __restrict__ A, const _Float16* __restrict__ Bp,
    float* __restrict__ C, const float* __restrict__ bias, float biasScale,
    int M, int N, int K, int lda, int ldc,
    long long strideA, long long strideC) {
  __shared__ __align__(32) _Float16 sA[2][4 * 32 * 16];  // ping-pong, frag order
  const int tid = threadIdx.x, lane = tid & 31, w = tid >> 5;
  const int n0 = blockIdx.x * 64, m0 = blockIdx.y * 64;
  const int ntN = N >> 4;
  const float* Ab = A + (long long)blockIdx.z * strideA;
  float* Cb = C + (long long)blockIdx.z * strideC;
  const int sr = w >> 1, sc0 = (w & 1) * 2;

  auto stage = [&](int buf, int k0) {
#pragma unroll
    for (int i = 0; i < 2; ++i) {
      int g = tid + i * 256;                    // 0..511
      int r = g >> 3, c4 = (g & 7) << 2;
      int gr = m0 + r;
      int rc = (gr < M) ? gr : (M - 1);
      float4 v = *(const float4*)(Ab + (long long)rc * lda + k0 + c4);
      if (gr >= M) { v.x = 0.f; v.y = 0.f; v.z = 0.f; v.w = 0.f; }
      int ln = (r & 15) | ((c4 & 8) << 1);
      int e = (c4 & 7) | ((c4 >> 1) & 8);
      v4h h = { (_Float16)v.x, (_Float16)v.y, (_Float16)v.z, (_Float16)v.w };
      *(v4h*)(sA[buf] + (((r >> 4) * 32 + ln) << 4) + e) = h;
    }
  };

  const int nks = K >> 5;
  stage(0, 0);
  v8f acc[2]; acc[0] = (v8f){}; acc[1] = (v8f){};
  for (int ks = 0; ks < nks; ++ks) {
    __syncthreads();                            // buf[ks&1] staging complete
    if (ks + 1 < nks) stage((ks + 1) & 1, (ks + 1) << 5);
    v16h a = *(const v16h*)(sA[ks & 1] + ((sr * 32 + lane) << 4));
    long long kb = (long long)ks * ntN;
#pragma unroll
    for (int j = 0; j < 2; ++j) {
      v16h b = *(const v16h*)(Bp + ((kb + (n0 >> 4) + sc0 + j) * 32 + lane) * 16);
      acc[j] = __builtin_amdgcn_wmma_f32_16x16x32_f16(false, a, false, b, (short)0,
                                                      acc[j], false, false);
    }
  }

  const int rowHi = (lane >> 4) << 3, colIn = lane & 15;
  const int gr0 = m0 + sr * 16 + rowHi;
  float cold[2][8];
  if (ACC) {                                    // branchless clause'd C loads
#pragma unroll
    for (int j = 0; j < 2; ++j) {
      int gc = n0 + (sc0 + j) * 16 + colIn;
#pragma unroll
      for (int r = 0; r < 8; ++r) {
        int gr = gr0 + r;
        int grc = (gr < M) ? gr : (M - 1);
        cold[j][r] = Cb[(long long)grc * ldc + gc];
      }
    }
  }
#pragma unroll
  for (int j = 0; j < 2; ++j) {
    int gc = n0 + (sc0 + j) * 16 + colIn;
    float bv = bias[gc] * biasScale;
#pragma unroll
    for (int r = 0; r < 8; ++r) {
      int gr = gr0 + r;
      float v = acc[j][r] + bv;
      if (GELU) v = 0.5f * v * (1.f + erff(v * 0.70710678118f));
      if (ACC) v += cold[j][r];
      if (gr < M) Cb[(long long)gr * ldc + gc] = v;
    }
  }
}

// ---------------- elevation indices + embedding gather ----------------
__global__ __launch_bounds__(256) void k_eleva(
    const float* __restrict__ xg, const float* __restrict__ per_li,
    const float* __restrict__ per_li_var, const float* __restrict__ ee,
    const float* __restrict__ ev, float* ele_vec, float* vele_vec) {
  __shared__ float rs[256], rss[256];
  __shared__ int sidx[2];
  int b = blockIdx.x, tid = threadIdx.x;
  const float* p = xg + (long long)b * NPX;
  float s = 0.f, ss = 0.f;
  for (int i = tid; i < NPX; i += 256) { float v = p[i]; s += v; ss += v * v; }
  rs[tid] = s; rss[tid] = ss; __syncthreads();
  for (int st = 128; st; st >>= 1) {
    if (tid < st) { rs[tid] += rs[tid + st]; rss[tid] += rss[tid + st]; }
    __syncthreads();
  }
  if (tid == 0) {
    float mean = rs[0] / (float)NPX;
    float var = (rss[0] - rs[0] * mean) / (float)(NPX - 1);   // ddof=1
    int i1 = 0, i2 = 0;
    for (int j = 0; j < 15; ++j) {
      i1 += (per_li[j] < mean);
      i2 += (per_li_var[j] < var);
    }
    sidx[0] = i1; sidx[1] = i2;
  }
  __syncthreads();
  if (tid < EMB) {
    ele_vec[b * EMB + tid]  = ee[sidx[0] * EMB + tid];
    vele_vec[b * EMB + tid] = ev[sidx[1] * EMB + tid];
  }
}

// ---------------- im2col ----------------
__global__ __launch_bounds__(256) void k_im2col(const float* __restrict__ x,
                                                float* __restrict__ out) {
  int blk = blockIdx.x;              // s*196 + p
  int s = blk / NPATCH, p = blk % NPATCH;
  int py = p / 14, px = p % 14;
#pragma unroll
  for (int j = 0; j < 3; ++j) {
    int idx = threadIdx.x + j * 256;          // 0..767 = c*256+dy*16+dx
    int c = idx >> 8, rem = idx & 255, dy = rem >> 4, dx = rem & 15;
    out[(long long)blk * 768 + idx] =
        x[(((long long)s * 3 + c) * IMGSZ + py * 16 + dy) * IMGSZ + px * 16 + dx];
  }
}
__global__ __launch_bounds__(256) void k_cls_pos(float* __restrict__ tok,
                                                 const float* __restrict__ cls,
                                                 const float* __restrict__ pos) {
  int s = blockIdx.x;
  for (int idx = threadIdx.x; idx < NTOK * EMB; idx += 256) {
    int t = idx / EMB, e = idx % EMB;
    long long a = (long long)s * NTOK * EMB + idx;
    if (t == 0) tok[a] = cls[e] + pos[idx];
    else        tok[a] += pos[idx];
  }
}

// ---------------- SAT ele/vele q,k,v vectors ----------------
__global__ __launch_bounds__(256) void k_sat_prep(
    const float* __restrict__ ele_vec, const float* __restrict__ vele_vec,
    const float* __restrict__ wq, const float* __restrict__ bq,
    const float* __restrict__ wk, const float* __restrict__ bk,
    const float* __restrict__ wv, const float* __restrict__ bv,
    float* __restrict__ satv) {
  int s = blockIdx.x, c = threadIdx.x;        // c < 256
  const float* e  = ele_vec  + (s % BTS) * EMB;
  const float* ve = vele_vec + (s % BTS) * EMB;
  float qe = bq[c], ke = bk[c], veq = bv[c];
  float qv = bq[c], kv = bk[c], vvv = bv[c];
  for (int k = 0; k < EMB; ++k) {
    float a = e[k], b = ve[k];
    float w0 = wq[k * SATD + c], w1 = wk[k * SATD + c], w2 = wv[k * SATD + c];
    qe += a * w0; ke += a * w1; veq += a * w2;
    qv += b * w0; kv += b * w1; vvv += b * w2;
  }
  long long base = (long long)s * 6 * SATD + c;
  satv[base + 0 * SATD] = qe;  satv[base + 1 * SATD] = ke;  satv[base + 2 * SATD] = veq;
  satv[base + 3 * SATD] = qv;  satv[base + 4 * SATD] = kv;  satv[base + 5 * SATD] = vvv;
}

// ---------------- SAT tiny 3-token attention (head dim 32 == wave32) ----------
__global__ __launch_bounds__(256) void k_sat_attn(
    const float* __restrict__ Qt, const float* __restrict__ Kt,
    const float* __restrict__ Vt, const float* __restrict__ satv,
    float* __restrict__ osum) {
  int gw = blockIdx.x * 8 + (threadIdx.x >> 5);
  if (gw >= NB * NTOK) return;
  int lane = threadIdx.x & 31;
  int s = gw / NTOK;
  const float* sv = satv + (long long)s * 6 * SATD;
  long long rb = (long long)gw * SATD;
  const float scale = 0.17677669529663687f;   // 1/sqrt(32)
#pragma unroll
  for (int h = 0; h < 8; ++h) {
    int d = h * 32 + lane;
    float q0 = Qt[rb + d], q1 = sv[0 * SATD + d], q2 = sv[3 * SATD + d];
    float k0 = Kt[rb + d], k1 = sv[1 * SATD + d], k2 = sv[4 * SATD + d];
    float v0 = Vt[rb + d], v1 = sv[2 * SATD + d], v2 = sv[5 * SATD + d];
    float p00 = wave_sum(q0 * k0), p01 = wave_sum(q0 * k1), p02 = wave_sum(q0 * k2);
    float p10 = wave_sum(q1 * k0), p11 = wave_sum(q1 * k1), p12 = wave_sum(q1 * k2);
    float p20 = wave_sum(q2 * k0), p21 = wave_sum(q2 * k1), p22 = wave_sum(q2 * k2);
    float o = 0.f;
#pragma unroll
    for (int q = 0; q < 3; ++q) {
      float a = (q == 0 ? p00 : q == 1 ? p10 : p20) * scale;
      float b = (q == 0 ? p01 : q == 1 ? p11 : p21) * scale;
      float c = (q == 0 ? p02 : q == 1 ? p12 : p22) * scale;
      float m = fmaxf(a, fmaxf(b, c));
      float e0 = __expf(a - m), e1 = __expf(b - m), e2 = __expf(c - m);
      float inv = 1.f / (e0 + e1 + e2);
      o += (e0 * v0 + e1 * v1 + e2 * v2) * inv;
    }
    osum[rb + d] = o;
  }
}

// ---------------- ViT attention, fused per (head, sample) in LDS -------------
// All WMMA operands in fragment order -> b128 LDS loads.
// LDS: S f32 [208x224] | {Q,K frag -> P frag} | V frag
#define S_ROWS 208
#define S_COLS 224
#define QF_H (13*2*32*16)   // 13312 halves
#define KF_H (14*2*32*16)   // 14336
#define PF_H (13*7*32*16)   // 46592
#define VF_H (7*4*32*16)    // 14336
#define ATT_LDS (S_ROWS*S_COLS*4 + PF_H*2 + VF_H*2)   // 308224 B
__global__ __launch_bounds__(256) void k_attn(const float* __restrict__ qkv,
                                              float* __restrict__ outp) {
  extern __shared__ __align__(32) char smem[];
  float* S = (float*)smem;
  _Float16* Pf = (_Float16*)(smem + S_ROWS * S_COLS * 4);
  _Float16* Qf = Pf;                  // Q, K live where P goes later
  _Float16* Kf = Pf + QF_H;
  _Float16* Vf = (_Float16*)(smem + S_ROWS * S_COLS * 4 + PF_H * 2);
  const int tid = threadIdx.x, lane = tid & 31, w = tid >> 5;
  const int h = blockIdx.x, s = blockIdx.y;
  const float* base = qkv + (long long)s * NTOK * 576;

  // stage Q (13 row-tiles) and K (14 col-tiles) in frag order, zero padded
  for (int idx = tid; idx < S_ROWS * 16; idx += 256) {
    int r = idx >> 4, d4 = (idx & 15) << 2;
    float4 v = {0.f, 0.f, 0.f, 0.f};
    if (r < NTOK) v = *(const float4*)(base + r * 576 + h * 64 + d4);
    int ks = d4 >> 5, kk = d4 & 31;
    int ln = (r & 15) | ((kk & 8) << 1);
    int e = (kk & 7) | ((kk >> 1) & 8);
    v4h hq = { (_Float16)v.x, (_Float16)v.y, (_Float16)v.z, (_Float16)v.w };
    *(v4h*)(Qf + ((((r >> 4) * 2 + ks) * 32 + ln) << 4) + e) = hq;
  }
  for (int idx = tid; idx < S_COLS * 16; idx += 256) {
    int r = idx >> 4, d4 = (idx & 15) << 2;
    float4 v = {0.f, 0.f, 0.f, 0.f};
    if (r < NTOK) v = *(const float4*)(base + r * 576 + EMB + h * 64 + d4);
    int ks = d4 >> 5, kk = d4 & 31;
    int ln = (r & 15) | ((kk & 8) << 1);
    int e = (kk & 7) | ((kk >> 1) & 8);
    v4h hk = { (_Float16)v.x, (_Float16)v.y, (_Float16)v.z, (_Float16)v.w };
    *(v4h*)(Kf + ((((r >> 4) * 2 + ks) * 32 + ln) << 4) + e) = hk;
  }
  // stage V (K-dim of O GEMM) in B-frag order (scatter across lanes)
  for (int idx = tid; idx < S_COLS * 16; idx += 256) {
    int r = idx >> 4, d4 = (idx & 15) << 2;
    float4 v = {0.f, 0.f, 0.f, 0.f};
    if (r < NTOK) v = *(const float4*)(base + r * 576 + 2 * EMB + h * 64 + d4);
    int ks = r >> 5, kk = r & 31;
    int e = (kk & 7) | ((kk >> 1) & 8);
    float vv[4] = {v.x, v.y, v.z, v.w};
#pragma unroll
    for (int j = 0; j < 4; ++j) {
      int d = d4 + j;
      int ln = (d & 15) | ((kk & 8) << 1);
      Vf[(((ks * 4 + (d >> 4)) * 32 + ln) << 4) + e] = (_Float16)vv[j];
    }
  }
  __syncthreads();

  // S = Q @ K^T * 1/8  (13x14 tiles, K=64)
  for (int t = w; t < 13 * 14; t += 8) {
    int mi = t / 14, ni = t % 14;
    v8f acc = (v8f){};
#pragma unroll
    for (int ks = 0; ks < 2; ++ks) {
      v16h a = *(const v16h*)(Qf + (((mi * 2 + ks) * 32 + lane) << 4));
      v16h b = *(const v16h*)(Kf + (((ni * 2 + ks) * 32 + lane) << 4));
      acc = __builtin_amdgcn_wmma_f32_16x16x32_f16(false, a, false, b, (short)0,
                                                   acc, false, false);
    }
    int rowHi = (lane >> 4) << 3, colIn = lane & 15;
#pragma unroll
    for (int r = 0; r < 8; ++r)
      S[(mi * 16 + r + rowHi) * S_COLS + ni * 16 + colIn] = acc[r] * 0.125f;
  }
  __syncthreads();

  // row softmax (cols >= 197 -> probability 0)
  for (int row = w; row < NTOK; row += 8) {
    float* Sr = S + row * S_COLS;
    float m = -1e30f;
    for (int c = lane; c < S_COLS; c += 32) if (c < NTOK) m = fmaxf(m, Sr[c]);
    m = wave_max(m);
    float sum = 0.f;
    for (int c = lane; c < S_COLS; c += 32) {
      if (c < NTOK) { float e = __expf(Sr[c] - m); Sr[c] = e; sum += e; }
      else Sr[c] = 0.f;
    }
    sum = wave_sum(sum);
    float inv = 1.f / sum;
    for (int c = lane; c < NTOK; c += 32) Sr[c] *= inv;
  }
  __syncthreads();

  // convert P to f16 fragment order (overwrites Q/K region)
  for (int idx = tid; idx < PF_H; idx += 256) {
    int e = idx & 15, ln = (idx >> 4) & 31, blk = idx >> 9;  // blk = mi*7+ks
    int ks = blk % 7, mi = blk / 7;
    int k = (e & 7) | ((ln & 16) >> 1) | ((e & 8) << 1);
    Pf[idx] = (_Float16)S[(mi * 16 + (ln & 15)) * S_COLS + ks * 32 + k];
  }
  __syncthreads();

  // O = P @ V  (13x4 tiles, K=224)
  for (int t = w; t < 13 * 4; t += 8) {
    int mi = t >> 2, ni = t & 3;
    v8f acc = (v8f){};
#pragma unroll
    for (int ks = 0; ks < 7; ++ks) {
      v16h a = *(const v16h*)(Pf + (((mi * 7 + ks) * 32 + lane) << 4));
      v16h b = *(const v16h*)(Vf + (((ks * 4 + ni) * 32 + lane) << 4));
      acc = __builtin_amdgcn_wmma_f32_16x16x32_f16(false, a, false, b, (short)0,
                                                   acc, false, false);
    }
    int rowHi = (lane >> 4) << 3, colIn = lane & 15;
#pragma unroll
    for (int r = 0; r < 8; ++r) {
      int row = mi * 16 + r + rowHi;
      if (row < NTOK)
        outp[((long long)s * NTOK + row) * SATD + h * 64 + ni * 16 + colIn] = acc[r];
    }
  }
}

// ---------------- LayerNorm: one wave32 per 192-wide row ----------------
__global__ __launch_bounds__(256) void k_ln(const float* __restrict__ x, int ldx,
                                            const float* __restrict__ w,
                                            const float* __restrict__ b,
                                            float* __restrict__ y, int ldy, int rows) {
  int gw = blockIdx.x * 8 + (threadIdx.x >> 5);
  if (gw >= rows) return;
  int lane = threadIdx.x & 31;
  const float* xr = x + (long long)gw * ldx;
  float v[6]; float s = 0.f;
#pragma unroll
  for (int i = 0; i < 6; ++i) { v[i] = xr[lane + 32 * i]; s += v[i]; }
  float mu = wave_sum(s) * (1.f / 192.f);
  float var = 0.f;
#pragma unroll
  for (int i = 0; i < 6; ++i) { float d = v[i] - mu; var += d * d; }
  var = wave_sum(var) * (1.f / 192.f);
  float inv = rsqrtf(var + 1e-5f);
  float* yr = y + (long long)gw * ldy;
#pragma unroll
  for (int i = 0; i < 6; ++i) {
    int c = lane + 32 * i;
    yr[c] = (v[i] - mu) * inv * w[c] + b[c];
  }
}

__global__ __launch_bounds__(256) void k_out(const float* __restrict__ act,
                                             float* __restrict__ out) {
  int idx = blockIdx.x * 256 + threadIdx.x;
  if (idx < NB * EMB) {
    int s = idx / EMB, e = idx % EMB;
    out[idx] = act[(long long)s * NTOK * SATD + e];
  }
}

// ---------------- host-side GEMM dispatch ----------------
static inline void launch_gemm(dim3 grid, hipStream_t st, int acc, int gelu,
                               const float* A, const _Float16* Bp, float* C,
                               const float* bias, float bs,
                               int M, int N, int K, int lda, int ldc,
                               long long sA, long long sC) {
  if (acc)
    k_gemm<1, 0><<<grid, 256, 0, st>>>(A, Bp, C, bias, bs, M, N, K, lda, ldc, sA, sC);
  else if (gelu)
    k_gemm<0, 1><<<grid, 256, 0, st>>>(A, Bp, C, bias, bs, M, N, K, lda, ldc, sA, sC);
  else
    k_gemm<0, 0><<<grid, 256, 0, st>>>(A, Bp, C, bias, bs, M, N, K, lda, ldc, sA, sC);
}

// ---------------- launcher ----------------
extern "C" void kernel_launch(void* const* d_in, const int* in_sizes, int n_in,
                              void* d_out, int out_size, void* d_ws, size_t ws_size,
                              hipStream_t stream) {
  (void)in_sizes; (void)n_in; (void)out_size; (void)ws_size;
  const float* x       = (const float*)d_in[0];
  const float* xg      = (const float*)d_in[1];
  const float* per_li  = (const float*)d_in[2];
  const float* per_liv = (const float*)d_in[3];
  const float* patch_w = (const float*)d_in[4];
  const float* patch_b = (const float*)d_in[5];
  const float* cls     = (const float*)d_in[6];
  const float* pos     = (const float*)d_in[7];
  const float* ee      = (const float*)d_in[8];
  const float* ev      = (const float*)d_in[9];
  const float* sat_wq  = (const float*)d_in[10];
  const float* sat_bq  = (const float*)d_in[11];
  const float* sat_wk  = (const float*)d_in[12];
  const float* sat_bk  = (const float*)d_in[13];
  const float* sat_wv  = (const float*)d_in[14];
  const float* sat_bv  = (const float*)d_in[15];
  const float* sat_wo  = (const float*)d_in[16];
  const float* sat_bo  = (const float*)d_in[17];
  const float* ln1_w   = (const float*)d_in[18];
  const float* ln1_b   = (const float*)d_in[19];
  const float* qkv_w   = (const float*)d_in[20];
  const float* proj_w  = (const float*)d_in[21];
  const float* proj_b  = (const float*)d_in[22];
  const float* ln2_w   = (const float*)d_in[23];
  const float* ln2_b   = (const float*)d_in[24];
  const float* fc1_w   = (const float*)d_in[25];
  const float* fc1_b   = (const float*)d_in[26];
  const float* fc2_w   = (const float*)d_in[27];
  const float* fc2_b   = (const float*)d_in[28];
  const float* lnf_w   = (const float*)d_in[29];
  const float* lnf_b   = (const float*)d_in[30];

  float* ws = (float*)d_ws;
  long long off = 0;
  float* tok  = ws + off;  off += (long long)NB * NTOK * EMB;
  float* act1 = ws + off;  off += (long long)NB * NTOK * SATD;
  float* big  = ws + off;  off += (long long)NB * NTOK * 768;
  float* elev = ws + off;  off += BTS * EMB;
  float* velev= ws + off;  off += BTS * EMB;
  float* satv = ws + off;  off += (long long)NB * 6 * SATD;

  // packed f16 weights (fragment order)
  _Float16* wp = (_Float16*)(ws + off);
  long long ho = 0;
  _Float16* pPatch = wp + ho; ho += 768 * EMB;
  _Float16* pSatQ  = wp + ho; ho += EMB * SATD;
  _Float16* pSatK  = wp + ho; ho += EMB * SATD;
  _Float16* pSatV  = wp + ho; ho += EMB * SATD;
  _Float16* pWo    = wp + ho; ho += SATD * EMB;
  _Float16* pQkv   = wp + ho; ho += (long long)VDEPTH * EMB * 576;
  _Float16* pProj  = wp + ho; ho += (long long)VDEPTH * EMB * EMB;
  _Float16* pFc1   = wp + ho; ho += (long long)VDEPTH * EMB * 768;
  _Float16* pFc2   = wp + ho; ho += (long long)VDEPTH * 768 * EMB;

  const long long SB = (long long)NB * NTOK * SATD;   // SAT q/k/v sub-buffer
  auto grd = [](long long n) { return (int)((n + 255) / 256); };

  // pack all weights (deterministic, graph-safe)
  k_packB<<<grd(768 * EMB), 256, 0, stream>>>(patch_w, pPatch, 768, EMB, 1);
  k_packB<<<grd(EMB * SATD), 256, 0, stream>>>(sat_wq, pSatQ, EMB, SATD, 0);
  k_packB<<<grd(EMB * SATD), 256, 0, stream>>>(sat_wk, pSatK, EMB, SATD, 0);
  k_packB<<<grd(EMB * SATD), 256, 0, stream>>>(sat_wv, pSatV, EMB, SATD, 0);
  k_packB<<<grd(SATD * EMB), 256, 0, stream>>>(sat_wo, pWo, SATD, EMB, 0);
  for (int i = 0; i < VDEPTH; ++i) {
    k_packB<<<grd(EMB * 576), 256, 0, stream>>>(qkv_w + (long long)i * EMB * 576,
                                                pQkv + (long long)i * EMB * 576, EMB, 576, 0);
    k_packB<<<grd(EMB * EMB), 256, 0, stream>>>(proj_w + (long long)i * EMB * EMB,
                                                pProj + (long long)i * EMB * EMB, EMB, EMB, 0);
    k_packB<<<grd(EMB * 768), 256, 0, stream>>>(fc1_w + (long long)i * EMB * 768,
                                                pFc1 + (long long)i * EMB * 768, EMB, 768, 0);
    k_packB<<<grd(768 * EMB), 256, 0, stream>>>(fc2_w + (long long)i * 768 * EMB,
                                                pFc2 + (long long)i * 768 * EMB, 768, EMB, 0);
  }

  // elevation embeddings + im2col
  k_eleva<<<BTS, 256, 0, stream>>>(xg, per_li, per_liv, ee, ev, elev, velev);
  k_im2col<<<NB * NPATCH, 256, 0, stream>>>(x, big);

  // patch embed GEMM -> tok rows 1..196, then cls + pos
  launch_gemm(dim3(EMB / 64, (NPATCH + 63) / 64, NB), stream, 0, 0,
              big, pPatch, tok + EMB, patch_b, 1.f, NPATCH, EMB, 768, 768, EMB,
              (long long)NPATCH * 768, (long long)NTOK * EMB);
  k_cls_pos<<<NB, 256, 0, stream>>>(tok, cls, pos);

  // SAT: tok-row q/k/v projections, ele/vele vectors, tiny attention, Wo
  launch_gemm(dim3(SATD / 64, 4, NB), stream, 0, 0,
              tok, pSatQ, big, sat_bq, 1.f, NTOK, SATD, EMB, EMB, SATD,
              (long long)NTOK * EMB, (long long)NTOK * SATD);
  launch_gemm(dim3(SATD / 64, 4, NB), stream, 0, 0,
              tok, pSatK, big + SB, sat_bk, 1.f, NTOK, SATD, EMB, EMB, SATD,
              (long long)NTOK * EMB, (long long)NTOK * SATD);
  launch_gemm(dim3(SATD / 64, 4, NB), stream, 0, 0,
              tok, pSatV, big + 2 * SB, sat_bv, 1.f, NTOK, SATD, EMB, EMB, SATD,
              (long long)NTOK * EMB, (long long)NTOK * SATD);
  k_sat_prep<<<NB, 256, 0, stream>>>(elev, velev, sat_wq, sat_bq, sat_wk, sat_bk,
                                     sat_wv, sat_bv, satv);
  k_sat_attn<<<(NB * NTOK) / 8, 256, 0, stream>>>(big, big + SB, big + 2 * SB,
                                                  satv, act1);
  // tok = osum @ Wo + 3*bo
  launch_gemm(dim3(EMB / 64, 4, NB), stream, 0, 0,
              act1, pWo, tok, sat_bo, 3.f, NTOK, EMB, SATD, SATD, EMB,
              (long long)NTOK * SATD, (long long)NTOK * EMB);

  // ViT blocks
  const int LNG = (NB * NTOK) / 8;
  for (int i = 0; i < VDEPTH; ++i) {
    k_ln<<<LNG, 256, 0, stream>>>(tok, EMB, ln1_w + i * EMB, ln1_b + i * EMB,
                                  act1, SATD, NB * NTOK);
    // qkv: no bias (scale 0, in-bounds dummy pointer of >=576 elems)
    launch_gemm(dim3(576 / 64, 4, NB), stream, 0, 0,
                act1, pQkv + (long long)i * EMB * 576, big, fc1_b, 0.f,
                NTOK, 576, EMB, SATD, 576,
                (long long)NTOK * SATD, (long long)NTOK * 576);
    k_attn<<<dim3(3, NB), 256, ATT_LDS, stream>>>(big, act1);
    launch_gemm(dim3(EMB / 64, 4, NB), stream, 1, 0,
                act1, pProj + (long long)i * EMB * EMB, tok, proj_b + i * EMB, 1.f,
                NTOK, EMB, EMB, SATD, EMB,
                (long long)NTOK * SATD, (long long)NTOK * EMB);
    k_ln<<<LNG, 256, 0, stream>>>(tok, EMB, ln2_w + i * EMB, ln2_b + i * EMB,
                                  act1, SATD, NB * NTOK);
    launch_gemm(dim3(768 / 64, 4, NB), stream, 0, 1,
                act1, pFc1 + (long long)i * EMB * 768, big, fc1_b + i * 768, 1.f,
                NTOK, 768, EMB, SATD, 768,
                (long long)NTOK * SATD, (long long)NTOK * 768);
    launch_gemm(dim3(EMB / 64, 4, NB), stream, 1, 0,
                big, pFc2 + (long long)i * 768 * EMB, tok, fc2_b + i * EMB, 1.f,
                NTOK, EMB, 768, 768, EMB,
                (long long)NTOK * 768, (long long)NTOK * EMB);
  }
  k_ln<<<LNG, 256, 0, stream>>>(tok, EMB, lnf_w, lnf_b, act1, SATD, NB * NTOK);
  k_out<<<(NB * EMB + 255) / 256, 256, 0, stream>>>(act1, (float*)d_out);
}
